// RoPETransformer_65034394796492
// MI455X (gfx1250) — matrix operand order repaired and checked by reading, not compiled, not run on hardware
//
#include <hip/hip_runtime.h>
#include <hip/hip_bf16.h>

// Problem constants (match reference)
#define DD    1024
#define HH    16
#define HDD   64
#define DFFD  4096
#define BB    32
#define SQn   200
#define SMEMn 1024

typedef __attribute__((ext_vector_type(16))) __bf16 v16bf;
typedef __attribute__((ext_vector_type(8)))  __bf16 v8bf;
typedef __attribute__((ext_vector_type(8)))  float  v8f;

union FragAB { v16bf v; v8bf h[2]; };

struct U2 { unsigned int x, y; };

// RNE fp32 -> bf16 round (result lives in high 16 bits)
static __device__ __forceinline__ unsigned int rne_hi(float a) {
  unsigned int u = __float_as_uint(a);
  return u + 0x7FFFu + ((u >> 16) & 1u);
}

// Pack two RNE-rounded bf16 into one dword with a single v_perm_b32.
static __device__ __forceinline__ unsigned int pack2bf(float a, float b) {
  return __builtin_amdgcn_perm(rne_hi(b), rne_hi(a), 0x07060302u);
}

static __device__ __forceinline__ U2 pack4bf(float4 f) {
  U2 r;
  r.x = pack2bf(f.x, f.y);
  r.y = pack2bf(f.z, f.w);
  return r;
}

static __device__ __forceinline__ unsigned short bf16bits(float a) {
  return (unsigned short)(rne_hi(a) >> 16);
}

// ---------------------------------------------------------------------------
// LayerNorm: one block per row of D=1024
// ---------------------------------------------------------------------------
__global__ __launch_bounds__(256)
void layernorm_kernel(const float* __restrict__ X, const float* __restrict__ g,
                      const float* __restrict__ bta, float* __restrict__ Y, int Mrows)
{
  __shared__ float red1[256];
  __shared__ float red2[256];
  const int row = blockIdx.x;
  if (row >= Mrows) return;
  const float* x = X + (size_t)row * DD;
  float xl[4], s1 = 0.f, s2 = 0.f;
#pragma unroll
  for (int i = 0; i < 4; ++i) {
    xl[i] = x[threadIdx.x + i * 256];
    s1 += xl[i];
    s2 += xl[i] * xl[i];
  }
  red1[threadIdx.x] = s1;
  red2[threadIdx.x] = s2;
  __syncthreads();
  for (int st = 128; st > 0; st >>= 1) {
    if ((int)threadIdx.x < st) {
      red1[threadIdx.x] += red1[threadIdx.x + st];
      red2[threadIdx.x] += red2[threadIdx.x + st];
    }
    __syncthreads();
  }
  const float mu  = red1[0] * (1.0f / DD);
  const float var = red2[0] * (1.0f / DD) - mu * mu;
  const float rs  = rsqrtf(var + 1e-5f);
  float* y = Y + (size_t)row * DD;
#pragma unroll
  for (int i = 0; i < 4; ++i) {
    const int c = threadIdx.x + i * 256;
    y[c] = (xl[i] - mu) * rs * g[c] + bta[c];
  }
}

// ---------------------------------------------------------------------------
// RoPE on [B*Sq, D] viewed as [tok, H, HD]; one thread per (even,odd) pair
// ---------------------------------------------------------------------------
__global__ __launch_bounds__(256)
void rope_kernel(const float* __restrict__ X, float* __restrict__ Y, int total)
{
  const int idx = blockIdx.x * 256 + threadIdx.x;
  if (idx >= total) return;
  const int tok = idx / (DD / 2);
  const int rem = idx % (DD / 2);
  const int hh  = rem / (HDD / 2);
  const int i   = rem % (HDD / 2);
  const int s   = tok % SQn;
  const float freq = __expf(-9.210340371976184f * (2.0f * (float)i) / (float)HDD);
  const float ang  = (float)s * freq;
  const float cs = __cosf(ang);
  const float sn = __sinf(ang);
  const size_t base = (size_t)tok * DD + (size_t)hh * HDD + 2 * i;
  const float x0 = X[base], x1 = X[base + 1];
  Y[base]     = x0 * cs - x1 * sn;
  Y[base + 1] = x1 * cs + x0 * sn;
}

// ---------------------------------------------------------------------------
// GEMM: C = act(A[M,K] @ W[N,K]^T + bias [+ residual]) via bf16 WMMA.
// 128x64 tile, BK=32, 256 threads = 8 waves (4 along M x 2 along N), each wave
// computes 32x32 (2x2 WMMA fragments). Double-buffered LDS: tile k+1's global
// loads are issued, tile k's WMMAs run while they are in flight, then pack +
// ds_store; one barrier per K step. K is a multiple of 32; rows clamped.
// vt_seq > 0: write output transposed per batch as C[b][n][s], s = m % vt_seq.
// ---------------------------------------------------------------------------
__global__ __launch_bounds__(256)
void gemm_bf16_wmma(const float* __restrict__ A, const float* __restrict__ W,
                    const float* __restrict__ bias, const float* __restrict__ residual,
                    float* __restrict__ C, int M, int N, int K, int relu, int vt_seq)
{
  __shared__ __align__(64) unsigned short sA[2][128 * 32];
  __shared__ __align__(64) unsigned short sB[2][64 * 32];
  const int t    = threadIdx.x;
  const int lane = t & 31, wid = t >> 5;
  const int wm = wid & 3, wn = wid >> 2;
  const int hi = lane >> 4, lo = lane & 15;
  const int m0 = blockIdx.y * 128, n0 = blockIdx.x * 64;

  const v8f vzero = {0.f, 0.f, 0.f, 0.f, 0.f, 0.f, 0.f, 0.f};
  v8f acc[4];
#pragma unroll
  for (int i = 0; i < 4; ++i) acc[i] = vzero;

  // Per-thread staging coordinates (float4 granularity, 8 float4 per 32-k row)
  const int srow = t >> 3;        // 0..31
  const int sc4  = t & 7;         // float4 column within the 32-wide K tile

  // Pre-clamped row base pointers (row clamp is safe for GEMM)
  const float* pA[4];
  const float* pW[2];
#pragma unroll
  for (int i = 0; i < 4; ++i) {
    int gm = m0 + srow + 32 * i; gm = gm < M ? gm : M - 1;
    pA[i] = A + (size_t)gm * K + 4 * sc4;
  }
#pragma unroll
  for (int i = 0; i < 2; ++i) {
    int gn = n0 + srow + 32 * i; gn = gn < N ? gn : N - 1;
    pW[i] = W + (size_t)gn * K + 4 * sc4;
  }

  float4 fA[4], fB[2];
  auto loadTile = [&](int k0) {
#pragma unroll
    for (int i = 0; i < 4; ++i) fA[i] = *(const float4*)(pA[i] + k0);
#pragma unroll
    for (int i = 0; i < 2; ++i) fB[i] = *(const float4*)(pW[i] + k0);
  };
  auto storeTile = [&](int buf) {
#pragma unroll
    for (int i = 0; i < 4; ++i)
      *(U2*)&sA[buf][(srow + 32 * i) * 32 + 4 * sc4] = pack4bf(fA[i]);
#pragma unroll
    for (int i = 0; i < 2; ++i)
      *(U2*)&sB[buf][(srow + 32 * i) * 32 + 4 * sc4] = pack4bf(fB[i]);
  };
  auto compute = [&](int buf) {
    const int kb = 16 * hi;
    v16bf b0 = *(const v16bf*)&sB[buf][(32 * wn + lo) * 32 + kb];
    v16bf b1 = *(const v16bf*)&sB[buf][(32 * wn + 16 + lo) * 32 + kb];
#pragma unroll
    for (int fa = 0; fa < 2; ++fa) {
      FragAB afr;
      const unsigned short* pa = &sA[buf][(32 * wm + 16 * fa + lo) * 32 + 16 * hi];
      afr.h[0] = *(const v8bf*)(pa);
      afr.h[1] = *(const v8bf*)(pa + 8);
      acc[fa * 2 + 0] = __builtin_amdgcn_wmma_f32_16x16x32_bf16(
          false, afr.v, false, b0, (short)0, acc[fa * 2 + 0], false, false);
      acc[fa * 2 + 1] = __builtin_amdgcn_wmma_f32_16x16x32_bf16(
          false, afr.v, false, b1, (short)0, acc[fa * 2 + 1], false, false);
    }
  };

  const int nk = K >> 5;  // K is a multiple of 32
  loadTile(0);
  storeTile(0);
  __syncthreads();
  for (int kt = 1; kt < nk; ++kt) {
    loadTile(kt * 32);                     // global loads in flight
    __builtin_prefetch(pA[0] + kt * 32 + 32, 0, 1);
    compute((kt - 1) & 1);                 // WMMAs on previous buffer
    storeTile(kt & 1);                     // pack + ds_store (consumes loads)
    __syncthreads();
  }
  compute((nk - 1) & 1);

  // Epilogue
#pragma unroll
  for (int fa = 0; fa < 2; ++fa) {
#pragma unroll
    for (int fb = 0; fb < 2; ++fb) {
      const v8f a = acc[fa * 2 + fb];
      const int n = n0 + 32 * wn + 16 * fb + lo;
      const float bz = bias ? bias[n] : 0.0f;
#pragma unroll
      for (int r = 0; r < 8; ++r) {
        const int m = m0 + 32 * wm + 16 * fa + r + 8 * hi;
        if (m >= M) continue;
        float v = a[r] + bz;
        if (vt_seq > 0) {
          const int b = m / vt_seq;
          const int s = m - b * vt_seq;
          C[((size_t)b * N + n) * vt_seq + s] = v;
        } else {
          if (residual) v += residual[(size_t)m * N + n];
          if (relu) v = v > 0.f ? v : 0.f;
          C[(size_t)m * N + n] = v;
        }
      }
    }
  }
}

// ---------------------------------------------------------------------------
// Flash attention (online softmax) with WMMA.
// One wave per (16-row Q tile, b*H + h). HD=64. Q/K are [B*S, D] with head
// columns h*HD..h*HD+63; V is pre-transposed: VT[b][hd(=D)][s] (leading Sk).
// 32-key tiles; 8 WMMAs per tile (4 QK^T + 4 PV). Sk is a multiple of 4, so
// per-float4 key validity is uniform -> mask packed dwords (2 cndmask).
// ---------------------------------------------------------------------------
__global__ __launch_bounds__(32)
void flash_attn_wmma(const float* __restrict__ Qg, const float* __restrict__ Kg,
                     const float* __restrict__ VTg, float* __restrict__ Og,
                     int Sq, int Sk, int causal)
{
  __shared__ __align__(64) unsigned short sQ[16 * 64];
  __shared__ __align__(64) unsigned short sK[32 * 64];
  __shared__ __align__(64) unsigned short sVt[64 * 32];  // [hd][key]
  __shared__ __align__(64) unsigned short sP[16 * 32];

  const int lane = threadIdx.x & 31;
  const int hi = lane >> 4, lo = lane & 15;
  const int qb = blockIdx.x * 16;
  const int bh = blockIdx.y;
  const int b = bh / HH, h = bh % HH;

  const float* q  = Qg  + ((size_t)b * Sq) * DD + h * HDD;
  const float* k  = Kg  + ((size_t)b * Sk) * DD + h * HDD;
  const float* vt = VTg + ((size_t)b * DD + h * HDD) * Sk;  // row stride Sk per hd
  float*       o  = Og  + ((size_t)b * Sq) * DD + h * HDD;

  // ---- Stage 16x64 Q tile (8 float4 per lane); dead rows masked in softmax ----
  {
    float4 f[8];
#pragma unroll
    for (int i = 0; i < 8; ++i) {
      const int idx4 = i * 32 + lane;          // 0..255
      const int r = idx4 >> 4, c4 = idx4 & 15; // row 0..15, float4 col 0..15
      int gq = qb + r; gq = gq < Sq ? gq : Sq - 1;
      f[i] = *(const float4*)(q + (size_t)gq * DD + 4 * c4);
    }
#pragma unroll
    for (int i = 0; i < 8; ++i) {
      const int idx4 = i * 32 + lane;
      const int r = idx4 >> 4, c4 = idx4 & 15;
      *(U2*)&sQ[r * 64 + 4 * c4] = pack4bf(f[i]);
    }
  }
  __syncthreads();

  FragAB aQ[2];
#pragma unroll
  for (int kk = 0; kk < 2; ++kk) {
    const unsigned short* pq = &sQ[lo * 64 + 32 * kk + 16 * hi];
    aQ[kk].h[0] = *(const v8bf*)(pq);
    aQ[kk].h[1] = *(const v8bf*)(pq + 8);
  }

  const v8f vzero = {0.f, 0.f, 0.f, 0.f, 0.f, 0.f, 0.f, 0.f};
  v8f accO[4];
#pragma unroll
  for (int f = 0; f < 4; ++f) accO[f] = vzero;
  float mrow[8], lrow[8];
#pragma unroll
  for (int r = 0; r < 8; ++r) { mrow[r] = -__builtin_inff(); lrow[r] = 0.f; }

  const float scale = 0.125f;  // 1/sqrt(64)

  for (int kt = 0; kt < Sk; kt += 32) {
    // ---- Stage 32x64 K tile (2 batches of 8 float4: load phase, then pack) ----
#pragma unroll
    for (int half = 0; half < 2; ++half) {
      float4 f[8];
      int valid[8];
#pragma unroll
      for (int i = 0; i < 8; ++i) {
        const int idx4 = (half * 8 + i) * 32 + lane;  // 0..511
        const int r = idx4 >> 4, c4 = idx4 & 15;      // key row 0..31
        const int gk = kt + r;
        const int gkc = gk < Sk ? gk : Sk - 1;
        valid[i] = gk < Sk;
        f[i] = *(const float4*)(k + (size_t)gkc * DD + 4 * c4);
      }
#pragma unroll
      for (int i = 0; i < 8; ++i) {
        const int idx4 = (half * 8 + i) * 32 + lane;
        const int r = idx4 >> 4, c4 = idx4 & 15;
        U2 p = pack4bf(f[i]);
        if (!valid[i]) { p.x = 0u; p.y = 0u; }
        *(U2*)&sK[r * 64 + 4 * c4] = p;
      }
    }
    // ---- Stage 64x32 V^T tile (2 batches of 8 float4) ----
#pragma unroll
    for (int half = 0; half < 2; ++half) {
      float4 f[8];
      int valid[8];
#pragma unroll
      for (int i = 0; i < 8; ++i) {
        const int idx4 = (half * 8 + i) * 32 + lane;  // 0..511
        const int hd = idx4 >> 3, c4 = idx4 & 7;      // hd row 0..63, key f4 0..7
        const int key0 = kt + 4 * c4;
        const int key0c = key0 < Sk ? key0 : Sk - 4;
        valid[i] = key0 < Sk;
        f[i] = *(const float4*)(vt + (size_t)hd * Sk + key0c);
      }
#pragma unroll
      for (int i = 0; i < 8; ++i) {
        const int idx4 = (half * 8 + i) * 32 + lane;
        const int hd = idx4 >> 3, c4 = idx4 & 7;
        U2 p = pack4bf(f[i]);
        if (!valid[i]) { p.x = 0u; p.y = 0u; }
        *(U2*)&sVt[hd * 32 + 4 * c4] = p;
      }
    }
    __syncthreads();

    // ---- S = Q @ K^T : two 16x16 fragments ----
    v8f s[2];
    s[0] = vzero; s[1] = vzero;
#pragma unroll
    for (int j = 0; j < 2; ++j) {
#pragma unroll
      for (int kk = 0; kk < 2; ++kk) {
        v16bf bk = *(const v16bf*)&sK[(16 * j + lo) * 64 + 32 * kk + 16 * hi];
        s[j] = __builtin_amdgcn_wmma_f32_16x16x32_bf16(false, aQ[kk].v, false, bk, (short)0, s[j], false, false);
      }
    }

    // ---- Online softmax; rows r + 8*hi live across the 16-lane groups ----
    float corr[8];
#pragma unroll
    for (int r = 0; r < 8; ++r) {
      const int grow = qb + r + 8 * hi;
      float sv[2];
      float mx = mrow[r];
#pragma unroll
      for (int j = 0; j < 2; ++j) {
        const int gcol = kt + 16 * j + lo;
        float x = s[j][r] * scale;
        const bool dead = (gcol >= Sk) || (grow >= Sq) || (causal && gcol > grow);
        x = dead ? -__builtin_inff() : x;
        sv[j] = x;
        mx = fmaxf(mx, x);
      }
#pragma unroll
      for (int d2 = 1; d2 < 16; d2 <<= 1)
        mx = fmaxf(mx, __shfl_xor(mx, d2, 32));
      const float c = (mrow[r] <= -1e30f) ? 0.f : __expf(mrow[r] - mx);
      corr[r] = c;
      float psum = 0.f;
#pragma unroll
      for (int j = 0; j < 2; ++j) {
        const float p = (sv[j] <= -1e30f) ? 0.f : __expf(sv[j] - mx);
        psum += p;
        sP[(r + 8 * hi) * 32 + 16 * j + lo] = bf16bits(p);
      }
#pragma unroll
      for (int d2 = 1; d2 < 16; d2 <<= 1)
        psum += __shfl_xor(psum, d2, 32);
      lrow[r] = lrow[r] * c + psum;
      mrow[r] = mx;
    }
    __syncthreads();

    // ---- Rescale running O, then O += P @ V ----
#pragma unroll
    for (int f = 0; f < 4; ++f)
#pragma unroll
      for (int r = 0; r < 8; ++r) accO[f][r] *= corr[r];

    FragAB aP;
    const unsigned short* pp = &sP[lo * 32 + 16 * hi];
    aP.h[0] = *(const v8bf*)(pp);
    aP.h[1] = *(const v8bf*)(pp + 8);
#pragma unroll
    for (int f = 0; f < 4; ++f) {
      v16bf bv = *(const v16bf*)&sVt[(16 * f + lo) * 32 + 16 * hi];
      accO[f] = __builtin_amdgcn_wmma_f32_16x16x32_bf16(false, aP.v, false, bv, (short)0, accO[f], false, false);
    }
    __syncthreads();
  }

  // ---- Normalize and store O ----
#pragma unroll
  for (int f = 0; f < 4; ++f) {
#pragma unroll
    for (int r = 0; r < 8; ++r) {
      const int grow = qb + r + 8 * hi;
      if (grow < Sq) {
        const float denom = lrow[r];
        const float val = (denom > 0.f) ? accO[f][r] / denom : 0.f;
        o[(size_t)grow * DD + 16 * f + lo] = val;
      }
    }
  }
}

// ---------------------------------------------------------------------------
// Host-side orchestration
// ---------------------------------------------------------------------------
extern "C" void kernel_launch(void* const* d_in, const int* in_sizes, int n_in,
                              void* d_out, int out_size, void* d_ws, size_t ws_size,
                              hipStream_t stream)
{
  (void)in_sizes; (void)n_in; (void)out_size; (void)ws_size;

  const float* tgt      = (const float*)d_in[0];
  const float* memory   = (const float*)d_in[1];
  const float* sa_in_w  = (const float*)d_in[2];
  const float* sa_in_b  = (const float*)d_in[3];
  const float* sa_out_w = (const float*)d_in[4];
  const float* sa_out_b = (const float*)d_in[5];
  const float* ca_in_w  = (const float*)d_in[6];
  const float* ca_in_b  = (const float*)d_in[7];
  const float* ca_out_w = (const float*)d_in[8];
  const float* ca_out_b = (const float*)d_in[9];
  const float* ff1_w    = (const float*)d_in[10];
  const float* ff1_b    = (const float*)d_in[11];
  const float* ff2_w    = (const float*)d_in[12];
  const float* ff2_b    = (const float*)d_in[13];
  const float* ln1_g    = (const float*)d_in[14];
  const float* ln1_b    = (const float*)d_in[15];
  const float* ln2_g    = (const float*)d_in[16];
  const float* ln2_b    = (const float*)d_in[17];
  const float* ln3_g    = (const float*)d_in[18];
  const float* ln3_b    = (const float*)d_in[19];

  float* out = (float*)d_out;
  float* ws  = (float*)d_ws;

  const size_t nTok  = (size_t)BB * SQn;     // 6400
  const size_t nTokD = nTok * DD;            // 6,553,600 floats
  const size_t nMem  = (size_t)BB * SMEMn;   // 32768
  const size_t nMemD = nMem * DD;            // 33,554,432 floats

  float* hbuf = ws;                  // LN outputs (h / h2 / h3)
  float* rop  = ws + nTokD;          // RoPE output
  float* qb   = ws + 2 * nTokD;      // Q buffer
  float* ob   = ws + 3 * nTokD;      // attention output buffer
  float* big  = ws + 4 * nTokD;      // self K,V^T | cross Kmem,V^Tmem | FFN hidden

  const dim3 blk256(256);
  const dim3 blk32(32);
  const dim3 gLN((unsigned)nTok);
  const dim3 gAttn((SQn + 15) / 16, BB * HH);
  auto gemmGrid = [](size_t M, int N) { return dim3((unsigned)((N + 63) / 64), (unsigned)((M + 127) / 128)); };

  // ---- Self-attention block ----
  layernorm_kernel<<<gLN, blk256, 0, stream>>>(tgt, ln1_g, ln1_b, hbuf, (int)nTok);

  const int ropeN = (int)(nTokD / 2);
  rope_kernel<<<dim3((ropeN + 255) / 256), blk256, 0, stream>>>(hbuf, rop, ropeN);

  // q,k from roped; v from h (v written transposed per batch: [b][D][Sq])
  gemm_bf16_wmma<<<gemmGrid(nTok, DD), blk256, 0, stream>>>(
      rop, sa_in_w, sa_in_b, nullptr, qb, (int)nTok, DD, DD, 0, 0);
  gemm_bf16_wmma<<<gemmGrid(nTok, DD), blk256, 0, stream>>>(
      rop, sa_in_w + (size_t)DD * DD, sa_in_b + DD, nullptr, big, (int)nTok, DD, DD, 0, 0);
  gemm_bf16_wmma<<<gemmGrid(nTok, DD), blk256, 0, stream>>>(
      hbuf, sa_in_w + 2 * (size_t)DD * DD, sa_in_b + 2 * DD, nullptr, big + nTokD,
      (int)nTok, DD, DD, 0, SQn);

  flash_attn_wmma<<<gAttn, blk32, 0, stream>>>(qb, big, big + nTokD, ob, SQn, SQn, 1);

  // x1 = tgt + o @ sa_out_w^T + b  -> d_out
  gemm_bf16_wmma<<<gemmGrid(nTok, DD), blk256, 0, stream>>>(
      ob, sa_out_w, sa_out_b, tgt, out, (int)nTok, DD, DD, 0, 0);

  // ---- Cross-attention block ----
  layernorm_kernel<<<gLN, blk256, 0, stream>>>(out, ln2_g, ln2_b, hbuf, (int)nTok);

  gemm_bf16_wmma<<<gemmGrid(nTok, DD), blk256, 0, stream>>>(
      hbuf, ca_in_w, ca_in_b, nullptr, qb, (int)nTok, DD, DD, 0, 0);
  gemm_bf16_wmma<<<gemmGrid(nMem, DD), blk256, 0, stream>>>(
      memory, ca_in_w + (size_t)DD * DD, ca_in_b + DD, nullptr, big, (int)nMem, DD, DD, 0, 0);
  gemm_bf16_wmma<<<gemmGrid(nMem, DD), blk256, 0, stream>>>(
      memory, ca_in_w + 2 * (size_t)DD * DD, ca_in_b + 2 * DD, nullptr, big + nMemD,
      (int)nMem, DD, DD, 0, SMEMn);

  flash_attn_wmma<<<gAttn, blk32, 0, stream>>>(qb, big, big + nMemD, ob, SQn, SMEMn, 0);

  // x2 = x1 + o @ ca_out_w^T + b  -> d_out (in-place residual)
  gemm_bf16_wmma<<<gemmGrid(nTok, DD), blk256, 0, stream>>>(
      ob, ca_out_w, ca_out_b, out, out, (int)nTok, DD, DD, 0, 0);

  // ---- FFN block ----
  layernorm_kernel<<<gLN, blk256, 0, stream>>>(out, ln3_g, ln3_b, hbuf, (int)nTok);

  gemm_bf16_wmma<<<gemmGrid(nTok, DFFD), blk256, 0, stream>>>(
      hbuf, ff1_w, ff1_b, nullptr, big, (int)nTok, DFFD, DD, 1, 0);
  gemm_bf16_wmma<<<gemmGrid(nTok, DD), blk256, 0, stream>>>(
      big, ff2_w, ff2_b, out, out, (int)nTok, DD, DFFD, 0, 0);
}
